// AttributeQuantizer_76493367542271
// MI455X (gfx1250) — compile-verified, matrix operand order checked by reading, not run on hardware
//
#include <hip/hip_runtime.h>
#include <hip/hip_bf16.h>

typedef __attribute__((ext_vector_type(16))) _Float16 v16h;
typedef __attribute__((ext_vector_type(8)))  float    v8f;

#define K_CODES 512
#define D_DIM   64
#define N_IN    262144
#define WAVES_PER_BLOCK 8
#define ROWS_PER_WAVE   16
#define NUM_WAVES (N_IN / ROWS_PER_WAVE)   // 16384

// ---------------------------------------------------------------------------
// Kernel 1: l2-normalize the codebook rows, store as f16 (row-major [K][D]).
// One wave (32 lanes) per row; each lane handles 2 columns.
// ---------------------------------------------------------------------------
__global__ void vq_norm_emb_kernel(const float* __restrict__ emb,
                                   _Float16* __restrict__ out) {
  const int row  = blockIdx.x;
  const int lane = threadIdx.x;          // 0..31
  const float2 v = ((const float2*)(emb + row * D_DIM))[lane];
  float ss = v.x * v.x + v.y * v.y;
  #pragma unroll
  for (int m = 16; m >= 1; m >>= 1) ss += __shfl_xor(ss, m, 32);
  const float inv = 1.0f / fmaxf(sqrtf(ss), 1e-12f);
  out[row * D_DIM + 2 * lane]     = (_Float16)(v.x * inv);
  out[row * D_DIM + 2 * lane + 1] = (_Float16)(v.y * inv);
}

// ---------------------------------------------------------------------------
// Kernel 2: init scalar outputs (perplexity = int32 1).
// ---------------------------------------------------------------------------
__global__ void vq_init_kernel(float* __restrict__ out) {
  ((int*)out)[1 + (size_t)N_IN * D_DIM] = 1;   // perplexity
}

// ---------------------------------------------------------------------------
// Kernel 3: main fused kernel.
//   - stage normalized codebook (64KB f16) into LDS
//   - normalize 16 input rows per wave into LDS (f16)
//   - WMMA f32_16x16x32_f16 over 32 N-tiles, fused running argmax
//   - write quantized gather, one-hot encodings, indices
//   - per-wave loss partials (reduced deterministically in kernel 4)
// ---------------------------------------------------------------------------
__launch_bounds__(256)
__global__ void vq_main_kernel(const float* __restrict__ x,
                               const int* __restrict__ labels,
                               const _Float16* __restrict__ embn,
                               const float* __restrict__ emb,
                               float* __restrict__ out,
                               float* __restrict__ loss_partials) {
  __shared__ _Float16 s_emb[K_CODES * D_DIM];                 // 64 KB
  __shared__ _Float16 s_a[WAVES_PER_BLOCK][ROWS_PER_WAVE * D_DIM]; // 16 KB

  const int tid  = threadIdx.x;
  const int wave = tid >> 5;
  const int lane = tid & 31;
  const int half = lane >> 4;    // 0: lanes 0-15, 1: lanes 16-31
  const int l16  = lane & 15;

  // ---- stage normalized codebook into LDS (uint4 = 8 halves per copy) ----
  {
    const uint4* src = (const uint4*)embn;
    uint4* dst = (uint4*)s_emb;
    #pragma unroll
    for (int i = tid; i < K_CODES * D_DIM / 8; i += 256) dst[i] = src[i];
  }
  __syncthreads();

  const int row0 = (blockIdx.x * WAVES_PER_BLOCK + wave) * ROWS_PER_WAVE;

  // ---- normalize this wave's 16 input rows into LDS as f16 ----
  _Float16* a_sh = s_a[wave];
  for (int r = 0; r < ROWS_PER_WAVE; ++r) {
    const float2 v = ((const float2*)(x + (size_t)(row0 + r) * D_DIM))[lane];
    float ss = v.x * v.x + v.y * v.y;
    #pragma unroll
    for (int m = 16; m >= 1; m >>= 1) ss += __shfl_xor(ss, m, 32);
    const float inv = 1.0f / fmaxf(sqrtf(ss), 1e-12f);
    a_sh[r * D_DIM + 2 * lane]     = (_Float16)(v.x * inv);
    a_sh[r * D_DIM + 2 * lane + 1] = (_Float16)(v.y * inv);
  }

  // ---- fragment packing per ISA 7.12.2 16-bit A layout (u32 = 2 halves) --
  union V16 { v16h h; unsigned int u[8]; };
  auto packA = [&](int k0) -> v16h {
    V16 r;
    const unsigned int* rowp = (const unsigned int*)(a_sh + l16 * D_DIM);
    const int kb = (k0 >> 1) + (half ? 4 : 0);
    #pragma unroll
    for (int t = 0; t < 4; ++t) {
      r.u[t]     = rowp[kb + t];       // K = k0 + halfsel + {0..7}
      r.u[4 + t] = rowp[kb + 8 + t];   // K = k0 + halfsel + {16..23}
    }
    return r.h;
  };
  auto packB = [&](int nt, int k0) -> v16h {   // B = E_n^T, column n per lane
    V16 r;
    const unsigned int* rowp =
        (const unsigned int*)(s_emb + (nt * 16 + l16) * D_DIM);
    const int kb = (k0 >> 1) + (half ? 4 : 0);
    #pragma unroll
    for (int t = 0; t < 4; ++t) {
      r.u[t]     = rowp[kb + t];
      r.u[4 + t] = rowp[kb + 8 + t];
    }
    return r.h;
  };

  const v16h a0 = packA(0);
  const v16h a1 = packA(32);

  float best[8];
  int   bidx[8];
  #pragma unroll
  for (int j = 0; j < 8; ++j) { best[j] = -3.0e38f; bidx[j] = 0; }

  // ---- GEMM tiles + fused argmax (first-max tie break via strict >) ----
  for (int nt = 0; nt < K_CODES / 16; ++nt) {
    v8f c = {};
    const v16h b0 = packB(nt, 0);
    const v16h b1 = packB(nt, 32);
    c = __builtin_amdgcn_wmma_f32_16x16x32_f16(false, a0, false, b0,
                                               (short)0, c, false, false);
    c = __builtin_amdgcn_wmma_f32_16x16x32_f16(false, a1, false, b1,
                                               (short)0, c, false, false);
    const int nidx = nt * 16 + l16;
    #pragma unroll
    for (int j = 0; j < 8; ++j) {
      const float v = c[j];               // row M = j + 8*half, col N = nidx
      if (v > best[j]) { best[j] = v; bidx[j] = nidx; }
    }
  }

  // ---- argmax butterfly within each 16-lane half (lower index on tie) ----
  #pragma unroll
  for (int j = 0; j < 8; ++j) {
    float v = best[j]; int idx = bidx[j];
    #pragma unroll
    for (int m = 8; m >= 1; m >>= 1) {
      const float ov = __shfl_xor(v, m, 32);
      const int   oi = __shfl_xor(idx, m, 32);
      if (ov > v || (ov == v && oi < idx)) { v = ov; idx = oi; }
    }
    best[j] = v; bidx[j] = idx;
  }

  // ---- output regions (tuple concatenated flat) ----
  float* q_out   = out + 1;                                         // [N,64]
  float* enc_out = out + 2 + (size_t)N_IN * D_DIM;                  // [N,512]
  int*   idx_out = (int*)(out + 2 + (size_t)N_IN * D_DIM
                              + (size_t)N_IN * K_CODES);            // [N]

  #pragma unroll
  for (int j = 0; j < 8; ++j) {
    const int row = row0 + j + 8 * half;
    const int idx = bidx[j];
    if (l16 == 0) idx_out[row] = idx;
    // quantized = embedding[idx] (raw, not normalized); 4 floats per lane
    const float* esrc = emb + (size_t)idx * D_DIM;
    float* qdst = q_out + (size_t)row * D_DIM;
    #pragma unroll
    for (int t = 0; t < 4; ++t) qdst[l16 * 4 + t] = esrc[l16 * 4 + t];
    // one-hot row: 32 cols per lane as float2 (base is 8B-aligned)
    float2* edst = (float2*)(enc_out + (size_t)row * K_CODES);
    #pragma unroll
    for (int t = 0; t < 16; ++t) {
      const int c0 = l16 * 32 + 2 * t;
      float2 v;
      v.x = (c0     == idx) ? 1.0f : 0.0f;
      v.y = (c0 + 1 == idx) ? 1.0f : 0.0f;
      edst[l16 * 16 + t] = v;
    }
  }

  // ---- commitment loss partial: sum over rows of (1 - <x_n, e_label>) ----
  float lsum = 0.0f;
  for (int r = 0; r < ROWS_PER_WAVE; ++r) {
    const int l = labels[row0 + r];
    const float ax0 = (float)a_sh[r * D_DIM + 2 * lane];
    const float ax1 = (float)a_sh[r * D_DIM + 2 * lane + 1];
    const float ex0 = (float)s_emb[l * D_DIM + 2 * lane];
    const float ex1 = (float)s_emb[l * D_DIM + 2 * lane + 1];
    float d = ax0 * ex0 + ax1 * ex1;
    #pragma unroll
    for (int m = 16; m >= 1; m >>= 1) d += __shfl_xor(d, m, 32);
    lsum += 1.0f - d;
  }
  if (lane == 0) loss_partials[blockIdx.x * WAVES_PER_BLOCK + wave] = lsum;
}

// ---------------------------------------------------------------------------
// Kernel 4: deterministic fixed-order loss reduction (no float atomics).
// ---------------------------------------------------------------------------
__global__ void vq_loss_reduce_kernel(const float* __restrict__ partials,
                                      float* __restrict__ out) {
  __shared__ float s[256];
  float acc = 0.0f;
  for (int i = threadIdx.x; i < NUM_WAVES; i += 256) acc += partials[i];
  s[threadIdx.x] = acc;
  __syncthreads();
  #pragma unroll
  for (int off = 128; off > 0; off >>= 1) {
    if (threadIdx.x < off) s[threadIdx.x] += s[threadIdx.x + off];
    __syncthreads();
  }
  if (threadIdx.x == 0) out[0] = s[0] * (1.0f / (float)N_IN);
}

// ---------------------------------------------------------------------------
extern "C" void kernel_launch(void* const* d_in, const int* in_sizes, int n_in,
                              void* d_out, int out_size, void* d_ws,
                              size_t ws_size, hipStream_t stream) {
  const float* x      = (const float*)d_in[0];   // [N, 64] f32
  const int*   labels = (const int*)d_in[1];     // [N] i32
  const float* emb    = (const float*)d_in[2];   // [512, 64] f32
  float* out = (float*)d_out;

  _Float16* embn = (_Float16*)d_ws;                               // 64 KB
  float* loss_partials = (float*)((char*)d_ws + 64 * 1024);       // 64 KB

  vq_init_kernel<<<1, 1, 0, stream>>>(out);
  vq_norm_emb_kernel<<<K_CODES, 32, 0, stream>>>(emb, embn);
  vq_main_kernel<<<N_IN / (WAVES_PER_BLOCK * ROWS_PER_WAVE), 256, 0, stream>>>(
      x, labels, embn, emb, out, loss_partials);
  vq_loss_reduce_kernel<<<1, 256, 0, stream>>>(loss_partials, out);
}
